// GraphMatchingUnifier_11553462026579
// MI455X (gfx1250) — matrix operand order repaired and checked by reading, not compiled, not run on hardware
//
#include <hip/hip_runtime.h>
#include <stdint.h>

#define D       128
#define NV      1024
#define NF      50000
#define NFP     50016            // NF padded to multiple of 32
#define NA      64
#define N_ITERS 3
#define SCALE_F 0.08838834764831845f   // 128^-0.5
#define INV_TAU 2.0f
#define NTILES_V (NV / 16)             // 64 var tiles

typedef __attribute__((ext_vector_type(16))) _Float16 v16h;
typedef __attribute__((ext_vector_type(8)))  _Float16 v8h;
typedef __attribute__((ext_vector_type(8)))  float    v8f;
typedef __attribute__((ext_vector_type(4)))  int      v4i;

// ---------------------------------------------------------------------------
// CDNA5 async global->LDS copies (guarded; falls back to plain staging)
// ---------------------------------------------------------------------------
#if defined(__AMDGCN__) && \
    __has_builtin(__builtin_amdgcn_global_load_async_to_lds_b128) && \
    __has_builtin(__builtin_amdgcn_s_wait_asynccnt)
#define USE_ASYNC_LDS 1
#else
#define USE_ASYNC_LDS 0
#endif

#if USE_ASYNC_LDS
typedef __attribute__((address_space(1))) v4i as1_v4i;
typedef __attribute__((address_space(3))) v4i as3_v4i;
__device__ __forceinline__ void async_copy_b128(const void* g, void* l) {
  // generic LDS pointer low 32 bits == LDS offset (ISA aperture scheme);
  // generic global pointer value == AS1 address. Integer casts are always
  // legal routes into the target address spaces.
  __builtin_amdgcn_global_load_async_to_lds_b128(
      (as1_v4i*)(unsigned long long)g,
      (as3_v4i*)(unsigned int)(unsigned long long)l, 0, 0);
}
#endif

__device__ __forceinline__ void stage_wait() {
#if USE_ASYNC_LDS
  __builtin_amdgcn_s_wait_asynccnt(0);
#endif
}

// Stage one F-chunk: K rows [fc, fc+32) x 128 (8 KB, contiguous) and
// V^T rows d=0..127 x cols [fc, fc+32) (8 KB, strided) into LDS.
// 256 threads x 4 x 16B copies. Buffer layout: K (4096 h) then V (4096 h).
__device__ __forceinline__ void stage_issue(const _Float16* __restrict__ Kh,
                                            const _Float16* __restrict__ VhT,
                                            int fc, _Float16* buf, int tid) {
  const _Float16* ksrc = Kh + (size_t)fc * D;
  _Float16* kbuf = buf;
  _Float16* vbuf = buf + 4096;
#pragma unroll
  for (int l = 0; l < 2; ++l) {
    int koff = tid * 16 + l * 8;       // halves
    int c = tid * 2 + l;               // V 16B-chunk id, 512 total
    int dv = c >> 2, part = c & 3;
    const _Float16* vsrc = VhT + (size_t)dv * NFP + fc + part * 8;
    _Float16* vdst = vbuf + dv * 32 + part * 8;
#if USE_ASYNC_LDS
    async_copy_b128(ksrc + koff, kbuf + koff);
    async_copy_b128(vsrc, vdst);
#else
    *reinterpret_cast<v8h*>(kbuf + koff) = *reinterpret_cast<const v8h*>(ksrc + koff);
    *reinterpret_cast<v8h*>(vdst) = *reinterpret_cast<const v8h*>(vsrc);
#endif
  }
}

// ---------------------------------------------------------------------------
// WMMA helpers (CDNA5 gfx1250, wave32)
// ---------------------------------------------------------------------------
__device__ __forceinline__ v8f wmma_f16(v16h a, v16h b, v8f c) {
  // D(16x16,f32) = A(16x32,f16) * B(32x16,f16) + C
  return __builtin_amdgcn_wmma_f32_16x16x32_f16(false, a, false, b,
                                                (short)0, c, false, false);
}

// A operand (16x32 f16), row-major source.
// Layout: lane m (<16): k = {0..7, 16..23}; lane m+16: k = {8..15, 24..31}.
__device__ __forceinline__ v16h load_tileA_h(const _Float16* __restrict__ base,
                                             int rowStride, int lane) {
  int r = lane & 15, hi = lane >> 4;
  const _Float16* p = base + (size_t)r * rowStride + hi * 8;
  v8h lo = *reinterpret_cast<const v8h*>(p);
  v8h hh = *reinterpret_cast<const v8h*>(p + 16);
  v16h out;
#pragma unroll
  for (int i = 0; i < 8; ++i) { out[i] = lo[i]; out[i + 8] = hh[i]; }
  return out;
}

// A operand from f32 row-major source, converting to f16.
__device__ __forceinline__ v16h load_tileA_f32(const float* __restrict__ base,
                                               int rowStride, int lane) {
  int r = lane & 15, hi = lane >> 4;
  const float* p = base + (size_t)r * rowStride + hi * 8;
  v16h out;
#pragma unroll
  for (int i = 0; i < 8; ++i) {
    out[i]     = (_Float16)p[i];
    out[i + 8] = (_Float16)p[16 + i];
  }
  return out;
}

// B operand (32x16 f16). B[k][n] = src[n][k], src row-major (rows = B columns).
// Layout: lane n (<16): k = {0..7, 16..23}; lane n+16: k = {8..15, 24..31}.
__device__ __forceinline__ v16h load_tileB_h(const _Float16* __restrict__ base,
                                             int rowStride, int lane) {
  int c = lane & 15, hi = lane >> 4;
  const _Float16* p = base + (size_t)c * rowStride + hi * 8;
  v8h lo = *reinterpret_cast<const v8h*>(p);
  v8h hh = *reinterpret_cast<const v8h*>(p + 16);
  v16h out;
#pragma unroll
  for (int i = 0; i < 8; ++i) { out[i] = lo[i]; out[i + 8] = hh[i]; }
  return out;
}

// ---------------------------------------------------------------------------
// Kernel 1: K = fe@Wk + bk (row-major f16), VhT = (fe@Wv + bv)^T (f16 [D][NFP])
// ---------------------------------------------------------------------------
__global__ __launch_bounds__(256) void kv_proj_kernel(
    const float* __restrict__ fe,
    const float* __restrict__ Wk, const float* __restrict__ bk,
    const float* __restrict__ Wv, const float* __restrict__ bv,
    _Float16* __restrict__ Kh, _Float16* __restrict__ VhT) {
  __shared__ __align__(32) _Float16 WkT[D * D];  // WkT[n*D + k] = Wk[k][n]
  __shared__ __align__(32) _Float16 WvT[D * D];
  int tid = threadIdx.x, lane = tid & 31, wave = tid >> 5;
  for (int i = tid; i < D * D; i += 256) {
    int k = i / D, n = i % D;
    WkT[n * D + k] = (_Float16)Wk[i];
    WvT[n * D + k] = (_Float16)Wv[i];
  }
  __syncthreads();

  int tile = blockIdx.x * 8 + wave;
  if (tile >= NFP / 16) return;
  int fbase = tile * 16;
  int r = lane & 15, hi = lane >> 4;

  if (fbase >= NF) {  // padding rows -> zeros (scores/ctx safe)
    for (int i = lane; i < 16 * D; i += 32) {
      int rr = i / D, dd = i % D;
      Kh[(size_t)(fbase + rr) * D + dd] = (_Float16)0.f;
      VhT[(size_t)dd * NFP + fbase + rr] = (_Float16)0.f;
    }
    return;
  }

  v16h aF[4];
#pragma unroll
  for (int kc = 0; kc < 4; ++kc)
    aF[kc] = load_tileA_f32(fe + (size_t)fbase * D + kc * 32, D, lane);

#pragma unroll
  for (int nb = 0; nb < 8; ++nb) {
    v8f accK = {}, accV = {};
#pragma unroll
    for (int kc = 0; kc < 4; ++kc) {
      v16h bK = load_tileB_h(&WkT[nb * 16 * D + kc * 32], D, lane);
      v16h bV = load_tileB_h(&WvT[nb * 16 * D + kc * 32], D, lane);
      accK = wmma_f16(aF[kc], bK, accK);
      accV = wmma_f16(aF[kc], bV, accV);
    }
    float bbk = bk[nb * 16 + r], bbv = bv[nb * 16 + r];
#pragma unroll
    for (int j = 0; j < 8; ++j) {
      int row = j + 8 * hi;          // C layout: lane = col n, vgpr j = row m
      int col = nb * 16 + r;
      Kh[(size_t)(fbase + row) * D + col] = (_Float16)(accK[j] + bbk);
      VhT[(size_t)col * NFP + fbase + row] = (_Float16)(accV[j] + bbv);
    }
  }
}

// ---------------------------------------------------------------------------
// Kernel 2: flash attention pass (one wave per 16-var tile; 8 waves/block).
// K/V chunks are staged block-wide into LDS (async double buffer).
// MODE 0: ctx = softmax(q K^T * s) @ V
// MODE 1: var_assign = softmax((S+g)/tau) @ V, + energy & entropy partials
// ---------------------------------------------------------------------------
template <int MODE>
__global__ __launch_bounds__(256) void attn_kernel(
    const float* __restrict__ h,
    const float* __restrict__ Wq, const float* __restrict__ bq,
    const _Float16* __restrict__ Kh, const _Float16* __restrict__ VhT,
    float* __restrict__ ctx,                   // MODE 0 output
    const float* __restrict__ gumbel_u,        // MODE 1 input
    float* __restrict__ var_assign,            // MODE 1 output
    float* __restrict__ epart, float* __restrict__ entpart) {
  // Phase A: smem = WqT[n*D+k].  Phase B (F loop): smem = K/V double buffer:
  //   buf b at smem + b*8192: K chunk 32x128 (4096 h) then V^T slice 128x32.
  __shared__ __align__(32) _Float16 smem[D * D];       // 32 KB, reused
  __shared__ __align__(32) _Float16 qlds[8][16 * D];   // 32 KB
  int tid = threadIdx.x, lane = tid & 31, wave = tid >> 5;
  for (int i = tid; i < D * D; i += 256) {
    int k = i / D, n = i % D;
    smem[n * D + k] = (_Float16)Wq[i];
  }
  __syncthreads();

  int tile = blockIdx.x * 8 + wave;
  int vbase = tile * 16;
  int r = lane & 15, hi = lane >> 4;

  // ---- q = h_tile @ Wq + bq, staged to LDS (row-major f16) ----
  v16h aH[4];
#pragma unroll
  for (int kc = 0; kc < 4; ++kc)
    aH[kc] = load_tileA_f32(h + (size_t)vbase * D + kc * 32, D, lane);
#pragma unroll
  for (int nb = 0; nb < 8; ++nb) {
    v8f acc = {};
#pragma unroll
    for (int kc = 0; kc < 4; ++kc) {
      v16h b = load_tileB_h(&smem[nb * 16 * D + kc * 32], D, lane);
      acc = wmma_f16(aH[kc], b, acc);
    }
    float bb = bq[nb * 16 + r];
#pragma unroll
    for (int j = 0; j < 8; ++j)
      qlds[wave][(j + 8 * hi) * D + nb * 16 + r] = (_Float16)(acc[j] + bb);
  }
  __syncthreads();   // qlds ready; WqT region dead after this point

  v16h bQ[4];  // q^T as B operand, resident across the whole F loop
#pragma unroll
  for (int kc = 0; kc < 4; ++kc)
    bQ[kc] = load_tileB_h(&qlds[wave][kc * 32], D, lane);

  v8f acc[8] = {};                      // 16 vars x 128 dims (C layout)
  float m_run = -1e30f, l_run = 0.f;    // plain-softmax stats (per var = lane)
  float m2 = -1e30f, l2 = 0.f;          // gumbel-softmax stats (MODE 1)
  float num2 = 0.f, nums = 0.f;         // sum w*s, sum p*s (MODE 1)

  stage_issue(Kh, VhT, 0, smem, tid);   // prologue: chunk 0 -> buffer 0
  const int NCH = NFP / 32;
  for (int ic = 0; ic < NCH; ++ic) {
    int fc = ic * 32;
    stage_wait();        // this wave's async copies into buf[ic&1] done
    __syncthreads();     // all waves' copies visible / reads of other buf done
    if (ic + 1 < NCH)
      stage_issue(Kh, VhT, fc + 32, smem + ((ic + 1) & 1) * 8192, tid);
    const _Float16* Ks = smem + (ic & 1) * 8192;
    const _Float16* Vs = Ks + 4096;

    // S^T tiles for F chunks [fc,fc+16) and [fc+16,fc+32): A = K rows, B = q^T
    v8f S0 = {}, S1 = {};
#pragma unroll
    for (int kc = 0; kc < 4; ++kc) {
      v16h aK0 = load_tileA_h(Ks + kc * 32, D, lane);
      v16h aK1 = load_tileA_h(Ks + 16 * D + kc * 32, D, lane);
      S0 = wmma_f16(aK0, bQ[kc], S0);
      S1 = wmma_f16(aK1, bQ[kc], S1);
    }
    float s[16];
#pragma unroll
    for (int j = 0; j < 8; ++j) { s[j] = S0[j] * SCALE_F; s[8 + j] = S1[j] * SCALE_F; }
    if (fc + 32 > NF) {  // tail masking (only padding region)
#pragma unroll
      for (int j = 0; j < 8; ++j) {
        if (fc + 8 * hi + j >= NF)      s[j]     = -1e30f;
        if (fc + 16 + 8 * hi + j >= NF) s[8 + j] = -1e30f;
      }
    }

    // plain-softmax online stats (needed by both modes)
    float lmax = s[0];
#pragma unroll
    for (int j = 1; j < 16; ++j) lmax = fmaxf(lmax, s[j]);
    lmax = fmaxf(lmax, __shfl_xor(lmax, 16));
    float newm = fmaxf(m_run, lmax);
    float esc = __expf(m_run - newm);

    if (MODE == 0) {
      float p[16], lsum = 0.f;
#pragma unroll
      for (int j = 0; j < 16; ++j) { p[j] = __expf(s[j] - newm); lsum += p[j]; }
      lsum += __shfl_xor(lsum, 16);
      l_run = l_run * esc + lsum;
      m_run = newm;
      v16h pA;  // C-layout of S^T chunk pair == A-layout of P
#pragma unroll
      for (int j = 0; j < 16; ++j) pA[j] = (_Float16)p[j];
#pragma unroll
      for (int db = 0; db < 8; ++db) {
        v16h bV = load_tileB_h(Vs + (db * 16) * 32, 32, lane);
        acc[db] = acc[db] * esc;
        acc[db] = wmma_f16(pA, bV, acc[db]);
      }
    } else {
      float lsum = 0.f, ssum = 0.f;
#pragma unroll
      for (int j = 0; j < 16; ++j) {
        float p = __expf(s[j] - newm);
        lsum += p; ssum += p * s[j];
      }
      lsum += __shfl_xor(lsum, 16);
      ssum += __shfl_xor(ssum, 16);
      l_run = l_run * esc + lsum;
      nums  = nums  * esc + ssum;
      m_run = newm;

      // gumbel-perturbed logits
      float t[16];
      size_t rowoff = (size_t)(vbase + r) * NF;
#pragma unroll
      for (int j = 0; j < 8; ++j) {
        int f0 = fc + 8 * hi + j;
        int f1 = fc + 16 + 8 * hi + j;
        int c0 = f0 < NF ? f0 : NF - 1;
        int c1 = f1 < NF ? f1 : NF - 1;
        float g0 = -__logf(-__logf(gumbel_u[rowoff + c0]));
        float g1 = -__logf(-__logf(gumbel_u[rowoff + c1]));
        t[j]     = (s[j]     + g0) * INV_TAU;
        t[8 + j] = (s[8 + j] + g1) * INV_TAU;
      }
      float tmax = t[0];
#pragma unroll
      for (int j = 1; j < 16; ++j) tmax = fmaxf(tmax, t[j]);
      tmax = fmaxf(tmax, __shfl_xor(tmax, 16));
      float newm2 = fmaxf(m2, tmax);
      float esc2 = __expf(m2 - newm2);
      float w[16], l2sum = 0.f, n2sum = 0.f;
#pragma unroll
      for (int j = 0; j < 16; ++j) {
        w[j] = __expf(t[j] - newm2);
        l2sum += w[j]; n2sum += w[j] * s[j];
      }
      l2sum += __shfl_xor(l2sum, 16);
      n2sum += __shfl_xor(n2sum, 16);
      l2 = l2 * esc2 + l2sum;
      num2 = num2 * esc2 + n2sum;
      m2 = newm2;
      v16h pA;
#pragma unroll
      for (int j = 0; j < 16; ++j) pA[j] = (_Float16)w[j];
#pragma unroll
      for (int db = 0; db < 8; ++db) {
        v16h bV = load_tileB_h(Vs + (db * 16) * 32, 32, lane);
        acc[db] = acc[db] * esc2;
        acc[db] = wmma_f16(pA, bV, acc[db]);
      }
    }
  }

  // ---- epilogue ----
  if (MODE == 0) {
#pragma unroll
    for (int j = 0; j < 8; ++j) {
      int row = j + 8 * hi;
      float inv = 1.0f / __shfl(l_run, row);  // l for var (vbase+row)
#pragma unroll
      for (int db = 0; db < 8; ++db)
        ctx[(size_t)(vbase + row) * D + db * 16 + r] = acc[db][j] * inv;
    }
  } else {
#pragma unroll
    for (int j = 0; j < 8; ++j) {
      int row = j + 8 * hi;
      float inv = 1.0f / __shfl(l2, row);
#pragma unroll
      for (int db = 0; db < 8; ++db)
        var_assign[(size_t)(vbase + row) * D + db * 16 + r] = acc[db][j] * inv;
    }
    // deterministic per-wave partials (lanes >=16 hold duplicates -> zero)
    float e  = (lane < 16) ? -(num2 / l2) : 0.f;
    float en = (lane < 16) ? (-(nums / l_run) + m_run + __logf(l_run)) : 0.f;
#pragma unroll
    for (int off = 16; off > 0; off >>= 1) {
      e  += __shfl_xor(e, off);
      en += __shfl_xor(en, off);
    }
    if (lane == 0) { epart[tile] = e; entpart[tile] = en; }
  }
}

// ---------------------------------------------------------------------------
// Gate FC1: mid = gelu([h , x2] @ W1 + b1). x2 broadcast for cooc (avg row).
// ---------------------------------------------------------------------------
__global__ __launch_bounds__(128) void gate_fc1_kernel(
    const float* __restrict__ h, const float* __restrict__ x2, int x2_bcast,
    const float* __restrict__ W1, const float* __restrict__ b1,
    float* __restrict__ mid) {
  int v = blockIdx.x, n = threadIdx.x;
  const float* hr = h + (size_t)v * D;
  const float* xr = x2 + (x2_bcast ? 0 : (size_t)v * D);
  float a = b1[n];
  for (int k = 0; k < D; ++k) a += hr[k] * W1[k * D + n];
  for (int k = 0; k < D; ++k) a += xr[k] * W1[(D + k) * D + n];
  mid[(size_t)v * D + n] = 0.5f * a * (1.0f + erff(a * 0.70710678118654752f));
}

// Main-loop FC2: g = sigmoid(mid@W2+b2); h = g*ctx + (1-g)*h
__global__ __launch_bounds__(128) void gate_fc2_main_kernel(
    const float* __restrict__ mid, const float* __restrict__ W2,
    const float* __restrict__ b2, const float* __restrict__ ctx,
    float* __restrict__ h) {
  int v = blockIdx.x, n = threadIdx.x;
  const float* mr = mid + (size_t)v * D;
  float a = b2[n];
  for (int k = 0; k < D; ++k) a += mr[k] * W2[k * D + n];
  float g = 1.0f / (1.0f + __expf(-a));
  float hv = h[(size_t)v * D + n];
  h[(size_t)v * D + n] = g * ctx[(size_t)v * D + n] + (1.0f - g) * hv;
}

// Cooc FC2: conditional blend toward the masked average.
__global__ __launch_bounds__(128) void gate_fc2_cooc_kernel(
    const float* __restrict__ mid, const float* __restrict__ W2,
    const float* __restrict__ b2, const float* __restrict__ avgcnt,
    const int* __restrict__ maskrow, float* __restrict__ h) {
  int v = blockIdx.x, n = threadIdx.x;
  const float* mr = mid + (size_t)v * D;
  float a = b2[n];
  for (int k = 0; k < D; ++k) a += mr[k] * W2[k * D + n];
  float g = 1.0f / (1.0f + __expf(-a));
  float av = avgcnt[n];
  float hv = h[(size_t)v * D + n];
  bool take = (avgcnt[D] >= 2.0f) && (maskrow[v] > 0);
  float upd = g * av + (1.0f - g) * hv;
  h[(size_t)v * D + n] = take ? upd : hv;
}

// Masked mean over vars for one atom. grid = D blocks.
__global__ __launch_bounds__(256) void cooc_avg_kernel(
    const float* __restrict__ h, const int* __restrict__ maskrow,
    float* __restrict__ avgcnt) {
  __shared__ float red[256];
  __shared__ float redc[256];
  int d = blockIdx.x, t = threadIdx.x;
  float s = 0.f, c = 0.f;
  for (int v = t; v < NV; v += 256) {
    float mv = (float)maskrow[v];
    s += mv * h[(size_t)v * D + d];
    c += mv;
  }
  red[t] = s; redc[t] = c;
  __syncthreads();
  for (int o = 128; o > 0; o >>= 1) {
    if (t < o) { red[t] += red[t + o]; redc[t] += redc[t + o]; }
    __syncthreads();
  }
  if (t == 0) {
    avgcnt[d] = red[0] / fmaxf(redc[0], 1.0f);
    if (d == 0) avgcnt[D] = redc[0];
  }
}

__global__ void copy_kernel(const float* __restrict__ src, float* __restrict__ dst, int n) {
  int i = blockIdx.x * blockDim.x + threadIdx.x;
  if (i < n) dst[i] = src[i];
}

__global__ void reduce_out_kernel(const float* __restrict__ epart,
                                  const float* __restrict__ entpart,
                                  float* __restrict__ energy,
                                  float* __restrict__ entropy) {
  if (threadIdx.x == 0 && blockIdx.x == 0) {
    float e = 0.f, en = 0.f;
    for (int i = 0; i < NTILES_V; ++i) { e += epart[i]; en += entpart[i]; }
    *energy = e;
    *entropy = en;
  }
}

// ---------------------------------------------------------------------------
extern "C" void kernel_launch(void* const* d_in, const int* in_sizes, int n_in,
                              void* d_out, int out_size, void* d_ws, size_t ws_size,
                              hipStream_t stream) {
  (void)in_sizes; (void)n_in; (void)out_size; (void)ws_size;
  const float* fe    = (const float*)d_in[0];
  const float* vinit = (const float*)d_in[1];
  const int*   cooc  = (const int*)d_in[2];
  const float* gum   = (const float*)d_in[3];
  const float* Wq = (const float*)d_in[4],  *bq = (const float*)d_in[5];
  const float* Wk = (const float*)d_in[6],  *bk = (const float*)d_in[7];
  const float* Wv = (const float*)d_in[8],  *bv = (const float*)d_in[9];
  const float* Wm1 = (const float*)d_in[10], *bm1 = (const float*)d_in[11];
  const float* Wm2 = (const float*)d_in[12], *bm2 = (const float*)d_in[13];
  const float* Wc1 = (const float*)d_in[14], *bc1 = (const float*)d_in[15];
  const float* Wc2 = (const float*)d_in[16], *bc2 = (const float*)d_in[17];

  char* ws = (char*)d_ws;
  _Float16* Kh  = (_Float16*)ws;  ws += (size_t)NFP * D * sizeof(_Float16);
  _Float16* VhT = (_Float16*)ws;  ws += (size_t)NFP * D * sizeof(_Float16);
  float* h      = (float*)ws;     ws += (size_t)NV * D * sizeof(float);
  float* ctx    = (float*)ws;     ws += (size_t)NV * D * sizeof(float);
  float* mid    = (float*)ws;     ws += (size_t)NV * D * sizeof(float);
  float* avgcnt = (float*)ws;     ws += (D + 1) * sizeof(float);
  float* epart  = (float*)ws;     ws += NTILES_V * sizeof(float);
  float* entpart= (float*)ws;     ws += NTILES_V * sizeof(float);

  float* out        = (float*)d_out;
  float* energy     = out;                 // [0]
  float* var_assign = out + 1;             // [1 .. NV*D]
  float* entropy    = out + 1 + NV * D;    // [last]

  // Precompute K (f16 row-major) and V^T (f16 [D][NFP]).
  kv_proj_kernel<<<(NFP / 16 + 7) / 8, 256, 0, stream>>>(fe, Wk, bk, Wv, bv, Kh, VhT);
  copy_kernel<<<(NV * D + 255) / 256, 256, 0, stream>>>(vinit, h, NV * D);

  for (int it = 0; it < N_ITERS; ++it) {
    attn_kernel<0><<<NTILES_V / 8, 256, 0, stream>>>(
        h, Wq, bq, Kh, VhT, ctx, nullptr, nullptr, nullptr, nullptr);
    gate_fc1_kernel<<<NV, 128, 0, stream>>>(h, ctx, 0, Wm1, bm1, mid);
    gate_fc2_main_kernel<<<NV, 128, 0, stream>>>(mid, Wm2, bm2, ctx, h);
    for (int a = 0; a < NA; ++a) {       // sequential scan over atoms
      const int* mrow = cooc + (size_t)a * NV;
      cooc_avg_kernel<<<D, 256, 0, stream>>>(h, mrow, avgcnt);
      gate_fc1_kernel<<<NV, 128, 0, stream>>>(h, avgcnt, 1, Wc1, bc1, mid);
      gate_fc2_cooc_kernel<<<NV, 128, 0, stream>>>(mid, Wc2, bc2, avgcnt, mrow, h);
    }
  }

  // Phase 3: gumbel-softmax substitution + energy/entropy.
  attn_kernel<1><<<NTILES_V / 8, 256, 0, stream>>>(
      h, Wq, bq, Kh, VhT, nullptr, gum, var_assign, epart, entpart);
  reduce_out_kernel<<<1, 32, 0, stream>>>(epart, entpart, energy, entropy);
}
